// LaneATTHead_80504866997036
// MI455X (gfx1250) — matrix operand order repaired and checked by reading, not compiled, not run on hardware
//
#include <hip/hip_runtime.h>
#include <hip/hip_bf16.h>

// ---------------- problem constants ----------------
constexpr int B_   = 8;
constexpr int CIN  = 256;
constexpr int CF   = 64;
constexpr int FH   = 22;
constexpr int FW   = 40;
constexpr int A_   = 2784;
constexpr int S_   = 72;
constexpr int NC   = 2;
constexpr int D_   = CF * FH;          // 1408
constexpr int AM1  = A_ - 1;           // 2783
constexpr int P_   = NC + 2 + 2 * S_;  // 148
constexpr int D2   = 2 * D_;           // 2816
constexpr int NH   = NC + 2 * S_;      // 146 head outputs

// padded strides (bf16 B-operand rows must be 16B aligned)
constexpr int ATWP = 2784;             // attn_w padded N stride (2783 -> 2784)
constexpr int HWS  = 160;              // headW padded N stride (146 -> 160)

// tiles
constexpr int MT1 = A_ / 16;                 // 174 M-tiles (exact)
constexpr int NT1 = (AM1 + 15) / 16;         // 174 N-tiles for scores
constexpr int NG1 = (NT1 + 3) / 4;           // 44 groups of 4 N-tiles
constexpr int NG2 = (D_ / 16) / 4;           // 22 groups of 4 N-tiles for att_feat
constexpr int MTH = (B_ * A_) / 16;          // 1392 M-tiles for heads (exact)
constexpr int NGH = 5;                       // 5 groups of 2 N-tiles (10 tiles, 160 cols)

typedef float  v8f  __attribute__((ext_vector_type(8)));
typedef __bf16 v8bf __attribute__((ext_vector_type(8)));
typedef __bf16 v16bf __attribute__((ext_vector_type(16)));

__device__ __forceinline__ v8f wmma_bf16(v16bf a, v16bf b, v8f c) {
    // D(16x16 f32) = A(16x32 bf16) * B(32x16 bf16) + C
    return __builtin_amdgcn_wmma_f32_16x16x32_bf16(
        false, a, false, b, (short)0, c, false, false);
}

// A fragment from stored bf16 (row-major, per-lane row m, base already at +half*8):
// elements 0..7 = K+0..7, elements 8..15 = K+16..23 (ISA 16-bit A 16x32 layout)
__device__ __forceinline__ v16bf loadA16(const __bf16* p) {
    v8bf a0 = *(const v8bf*)(p);
    v8bf a1 = *(const v8bf*)(p + 16);
    return __builtin_shufflevector(a0, a1, 0,1,2,3,4,5,6,7,8,9,10,11,12,13,14,15);
}

// B fragment from stored bf16: per-lane column n, base already at row +half*8;
// elements 0..7 = K+0..7, elements 8..15 = K+16..23 (row stride ld)
__device__ __forceinline__ v16bf loadB16(const __bf16* p, int ld) {
    v16bf b;
    #pragma unroll
    for (int i = 0; i < 8; ++i) {
        b[i] = p[(size_t)i * ld];
        b[8 + i] = p[(size_t)(16 + i) * ld];
    }
    return b;
}

// on-the-fly hi/lo split of an f32 A row chunk (16 elements: +0..7 and +16..23)
__device__ __forceinline__ void splitA(const float* p, v16bf& h, v16bf& l) {
    v8f x0 = *(const v8f*)(p);
    v8f x1 = *(const v8f*)(p + 16);
    #pragma unroll
    for (int i = 0; i < 8; ++i) {
        __bf16 h0 = (__bf16)x0[i];
        h[i] = h0;       l[i] = (__bf16)(x0[i] - (float)h0);
        __bf16 h1 = (__bf16)x1[i];
        h[8 + i] = h1;   l[8 + i] = (__bf16)(x1[i] - (float)h1);
    }
}

// ---------------- kernel 1: 1x1 conv ----------------
__global__ __launch_bounds__(256) void k_conv1x1(
    const float* __restrict__ x, const float* __restrict__ conv_w,
    const float* __restrict__ conv_b, float* __restrict__ feats) {
    int idx = blockIdx.x * 256 + threadIdx.x;          // over B*CF*FH*FW
    int hw  = idx % (FH * FW);
    int o   = (idx / (FH * FW)) % CF;
    int b   = idx / (CF * FH * FW);
    const float* xp = x + (size_t)b * CIN * FH * FW + hw;
    const float* wp = conv_w + (size_t)o * CIN;
    float acc = conv_b[o];
    #pragma unroll 8
    for (int c = 0; c < CIN; ++c) acc = fmaf(xp[(size_t)c * FH * FW], wp[c], acc);
    feats[idx] = acc;
}

// ---------------- kernel 2a: split attn_w into padded bf16 hi/lo ----------------
__global__ __launch_bounds__(256) void k_split_attn_w(
    const float* __restrict__ attn_w, __bf16* __restrict__ wh, __bf16* __restrict__ wl) {
    int idx = blockIdx.x * 256 + threadIdx.x;          // over D_*ATWP
    int n = idx % ATWP, kd = idx / ATWP;
    float f = (n < AM1) ? attn_w[(size_t)kd * AM1 + n] : 0.0f;
    __bf16 h = (__bf16)f;
    wh[idx] = h;
    wl[idx] = (__bf16)(f - (float)h);
}

// ---------------- kernel 2b: pack + split head weights ----------------
__global__ __launch_bounds__(256) void k_prep_head(
    const float* __restrict__ cls_w, const float* __restrict__ cls_b,
    const float* __restrict__ reg_w, const float* __restrict__ reg_b,
    __bf16* __restrict__ wh, __bf16* __restrict__ wl, float* __restrict__ headB) {
    int idx = blockIdx.x * 256 + threadIdx.x;          // over D2*HWS
    int j = idx % HWS, d = idx / HWS;
    float f = 0.0f;
    if (j < NC)      f = cls_w[d * NC + j];
    else if (j < NH) f = reg_w[d * (2 * S_) + (j - NC)];
    __bf16 h = (__bf16)f;
    wh[idx] = h;
    wl[idx] = (__bf16)(f - (float)h);
    if (idx < NH) headB[idx] = (idx < NC) ? cls_b[idx] : reg_b[idx - NC];
}

// ---------------- kernel 3: gather + mask -> baf hi/lo (B,A,D) ----------------
__global__ __launch_bounds__(256) void k_gather(
    const float* __restrict__ feats, const int* __restrict__ cut_xs,
    const unsigned char* __restrict__ invalid_mask,
    __bf16* __restrict__ bh, __bf16* __restrict__ bl) {
    size_t idx = (size_t)blockIdx.x * 256 + threadIdx.x;  // over B*A*D
    int d = (int)(idx % D_);
    int a = (int)((idx / D_) % A_);
    int b = (int)(idx / ((size_t)A_ * D_));
    int o = d / FH, h = d % FH;
    float v = 0.0f;
    if (!invalid_mask[a * FH + h]) {
        int w = cut_xs[a * FH + h];
        v = feats[(((size_t)b * CF + o) * FH + h) * FW + w];
    }
    __bf16 vh = (__bf16)v;
    bh[idx] = vh;
    bl[idx] = (__bf16)(v - (float)vh);
}

// ---------------- kernel 4: scores GEMM (split-bf16) -> scattered into attn ----------------
__global__ __launch_bounds__(256) void k_gemm_scores(
    const __bf16* __restrict__ bafh, const __bf16* __restrict__ bafl,
    const __bf16* __restrict__ awh, const __bf16* __restrict__ awl,
    const float* __restrict__ attn_b, float* __restrict__ attn) {
    int wid  = blockIdx.x * 8 + (threadIdx.x >> 5);
    int lane = threadIdx.x & 31;
    int half = lane >> 4, l15 = lane & 15;
    if (wid >= B_ * MT1 * NG1) return;
    int b   = wid / (MT1 * NG1);
    int rem = wid % (MT1 * NG1);
    int m0  = (rem / NG1) * 16;
    int n0  = (rem % NG1) * 64;

    const __bf16* Ah = bafh + (size_t)b * A_ * D_ + (size_t)(m0 + l15) * D_ + half * 8;
    const __bf16* Al = bafl + (size_t)b * A_ * D_ + (size_t)(m0 + l15) * D_ + half * 8;
    const __bf16* Bh = awh + (size_t)(half * 8) * ATWP + n0 + l15;   // lane = N column
    const __bf16* Bl = awl + (size_t)(half * 8) * ATWP + n0 + l15;

    v8f cc[4] = {v8f{}, v8f{}, v8f{}, v8f{}};
    for (int k = 0; k < D_; k += 32) {
        v16bf ah = loadA16(Ah + k);
        v16bf al = loadA16(Al + k);
        const __bf16* bhp = Bh + (size_t)k * ATWP;
        const __bf16* blp = Bl + (size_t)k * ATWP;
        #pragma unroll
        for (int t = 0; t < 4; ++t) {
            v16bf bh = loadB16(bhp + t * 16, ATWP);
            v16bf bl = loadB16(blp + t * 16, ATWP);
            cc[t] = wmma_bf16(al, bh, cc[t]);
            cc[t] = wmma_bf16(ah, bl, cc[t]);
            cc[t] = wmma_bf16(ah, bh, cc[t]);
        }
    }
    int mbase = m0 + half * 8;
    size_t rowbase = ((size_t)b * A_ + mbase) * A_;
    #pragma unroll
    for (int t = 0; t < 4; ++t) {
        int n = n0 + t * 16 + l15;
        if (n >= AM1) continue;
        float bias = attn_b[n];
        #pragma unroll
        for (int r = 0; r < 8; ++r) {
            int m = mbase + r;
            int col = n + (n >= m ? 1 : 0);
            attn[rowbase + (size_t)r * A_ + col] = cc[t][r] + bias;
        }
    }
}

// ---------------- kernel 5: row softmax in-place on attn (diag -> 0) ----------------
__global__ __launch_bounds__(256) void k_softmax(float* __restrict__ attn) {
    __shared__ float buf[A_];
    __shared__ float red[256];
    int row = blockIdx.x;                 // 0 .. B*A-1
    int a   = row % A_;
    float* p = attn + (size_t)row * A_;
    int t = threadIdx.x;

    float mx = -3.0e38f;
    for (int c = t; c < A_; c += 256) {
        float v = (c == a) ? -3.0e38f : p[c];
        buf[c] = v;
        mx = fmaxf(mx, v);
    }
    red[t] = mx; __syncthreads();
    for (int s = 128; s > 0; s >>= 1) { if (t < s) red[t] = fmaxf(red[t], red[t + s]); __syncthreads(); }
    mx = red[0]; __syncthreads();

    float sum = 0.f;
    for (int c = t; c < A_; c += 256) {
        float e = (c == a) ? 0.f : expf(buf[c] - mx);
        buf[c] = e;
        sum += e;
    }
    red[t] = sum; __syncthreads();
    for (int s = 128; s > 0; s >>= 1) { if (t < s) red[t] += red[t + s]; __syncthreads(); }
    float inv = 1.0f / red[0];
    for (int c = t; c < A_; c += 256) p[c] = buf[c] * inv;
}

// ---------------- kernel 6: att_feat = attn @ baf (A split on the fly) ----------------
__global__ __launch_bounds__(256) void k_gemm_att(
    const float* __restrict__ attn,
    const __bf16* __restrict__ bafh, const __bf16* __restrict__ bafl,
    float* __restrict__ att_feat) {
    int wid  = blockIdx.x * 8 + (threadIdx.x >> 5);
    int lane = threadIdx.x & 31;
    int half = lane >> 4, l15 = lane & 15;
    if (wid >= B_ * MT1 * NG2) return;
    int b   = wid / (MT1 * NG2);
    int rem = wid % (MT1 * NG2);
    int m0  = (rem / NG2) * 16;
    int n0  = (rem % NG2) * 64;

    const float*  Ap = attn + (size_t)b * A_ * A_ + (size_t)(m0 + l15) * A_ + half * 8;
    const __bf16* Bh = bafh + (size_t)b * A_ * D_ + (size_t)(half * 8) * D_ + n0 + l15;
    const __bf16* Bl = bafl + (size_t)b * A_ * D_ + (size_t)(half * 8) * D_ + n0 + l15;

    v8f cc[4] = {v8f{}, v8f{}, v8f{}, v8f{}};
    for (int k = 0; k < A_; k += 32) {
        v16bf ah, al;
        splitA(Ap + k, ah, al);
        const __bf16* bhp = Bh + (size_t)k * D_;
        const __bf16* blp = Bl + (size_t)k * D_;
        #pragma unroll
        for (int t = 0; t < 4; ++t) {
            v16bf bh = loadB16(bhp + t * 16, D_);
            v16bf bl = loadB16(blp + t * 16, D_);
            cc[t] = wmma_bf16(al, bh, cc[t]);
            cc[t] = wmma_bf16(ah, bl, cc[t]);
            cc[t] = wmma_bf16(ah, bh, cc[t]);
        }
    }
    int mbase = m0 + half * 8;
    size_t outbase = ((size_t)b * A_ + mbase) * D_;
    #pragma unroll
    for (int t = 0; t < 4; ++t) {
        #pragma unroll
        for (int r = 0; r < 8; ++r)
            att_feat[outbase + (size_t)r * D_ + n0 + t * 16 + l15] = cc[t][r];
    }
}

// ---------------- kernel 7: head GEMM over f2 = [att_feat | baf] ----------------
__global__ __launch_bounds__(256) void k_gemm_head(
    const float* __restrict__ att_feat,
    const __bf16* __restrict__ bafh, const __bf16* __restrict__ bafl,
    const __bf16* __restrict__ hwh, const __bf16* __restrict__ hwl,
    const float* __restrict__ headB, float* __restrict__ head_out) {
    int wid  = blockIdx.x * 8 + (threadIdx.x >> 5);
    int lane = threadIdx.x & 31;
    int half = lane >> 4, l15 = lane & 15;
    if (wid >= MTH * NGH) return;
    int m0 = (wid / NGH) * 16;     // global row tile over B*A
    int n0 = (wid % NGH) * 32;     // 2 N-tiles

    const float*  Af = att_feat + (size_t)(m0 + l15) * D_ + half * 8;
    const __bf16* Ah = bafh + (size_t)(m0 + l15) * D_ + half * 8;
    const __bf16* Al = bafl + (size_t)(m0 + l15) * D_ + half * 8;
    const __bf16* Bh = hwh + (size_t)(half * 8) * HWS + n0 + l15;
    const __bf16* Bl = hwl + (size_t)(half * 8) * HWS + n0 + l15;

    v8f cc[2] = {v8f{}, v8f{}};
    // phase 1: k in [0, D_) from att_feat (f32, split on the fly)
    for (int k = 0; k < D_; k += 32) {
        v16bf ah, al;
        splitA(Af + k, ah, al);
        const __bf16* bhp = Bh + (size_t)k * HWS;
        const __bf16* blp = Bl + (size_t)k * HWS;
        #pragma unroll
        for (int t = 0; t < 2; ++t) {
            v16bf bh = loadB16(bhp + t * 16, HWS);
            v16bf bl = loadB16(blp + t * 16, HWS);
            cc[t] = wmma_bf16(al, bh, cc[t]);
            cc[t] = wmma_bf16(ah, bl, cc[t]);
            cc[t] = wmma_bf16(ah, bh, cc[t]);
        }
    }
    // phase 2: k in [D_, D2) from stored baf hi/lo
    for (int k = D_; k < D2; k += 32) {
        v16bf ah = loadA16(Ah + (k - D_));
        v16bf al = loadA16(Al + (k - D_));
        const __bf16* bhp = Bh + (size_t)k * HWS;
        const __bf16* blp = Bl + (size_t)k * HWS;
        #pragma unroll
        for (int t = 0; t < 2; ++t) {
            v16bf bh = loadB16(bhp + t * 16, HWS);
            v16bf bl = loadB16(blp + t * 16, HWS);
            cc[t] = wmma_bf16(al, bh, cc[t]);
            cc[t] = wmma_bf16(ah, bl, cc[t]);
            cc[t] = wmma_bf16(ah, bh, cc[t]);
        }
    }
    int mbase = m0 + half * 8;
    #pragma unroll
    for (int t = 0; t < 2; ++t) {
        int n = n0 + t * 16 + l15;
        if (n >= NH) continue;
        float bias = headB[n];
        #pragma unroll
        for (int r = 0; r < 8; ++r)
            head_out[(size_t)(mbase + r) * NH + n] = cc[t][r] + bias;
    }
}

// ---------------- kernel 8: assemble props ----------------
__global__ __launch_bounds__(256) void k_props(
    const float* __restrict__ head_out, const float* __restrict__ anchors,
    float* __restrict__ props) {
    int idx = blockIdx.x * 256 + threadIdx.x;      // over B*A*P
    int p = idx % P_;
    int r = idx / P_;
    int a = r % A_;
    float v;
    if (p < NC) {
        v = head_out[(size_t)r * NH + p];
    } else if (p < NC + 2) {
        v = anchors[(size_t)a * P_ + p];
    } else {
        int j = p - (NC + 2);                      // 0..143
        float g = head_out[(size_t)r * NH + NC + j];
        if (j >= S_) g = 1.0f / (1.0f + expf(-g));
        v = anchors[(size_t)a * P_ + p] + g;
    }
    props[idx] = v;
}

// ---------------- launcher ----------------
extern "C" void kernel_launch(void* const* d_in, const int* in_sizes, int n_in,
                              void* d_out, int out_size, void* d_ws, size_t ws_size,
                              hipStream_t stream) {
    const float* x        = (const float*)d_in[0];
    const float* conv_w   = (const float*)d_in[1];
    const float* conv_b   = (const float*)d_in[2];
    const float* attn_w   = (const float*)d_in[3];
    const float* attn_b   = (const float*)d_in[4];
    const float* cls_w    = (const float*)d_in[5];
    const float* cls_b    = (const float*)d_in[6];
    const float* reg_w    = (const float*)d_in[7];
    const float* reg_b    = (const float*)d_in[8];
    const float* anchors  = (const float*)d_in[9];
    const int*   cut_xs   = (const int*)d_in[10];
    const unsigned char* invalid_mask = (const unsigned char*)d_in[11];

    float* out   = (float*)d_out;
    float* props = out;                                  // B*A*P
    float* attn  = out + (size_t)B_ * A_ * P_;           // B*A*A (f32, also GEMM2 A-operand)

    // workspace carve (all offsets 16B aligned)
    float* feats    = (float*)d_ws;                      // 450560 f
    float* att_feat = feats    + (size_t)B_ * CF * FH * FW;   // 31358976 f
    float* head_out = att_feat + (size_t)B_ * A_ * D_;        // 3251712 f
    float* headB    = head_out + (size_t)B_ * A_ * NH;        // 160 f reserved
    __bf16* bafh    = (__bf16*)(headB + 160);
    __bf16* bafl    = bafh + (size_t)B_ * A_ * D_;
    __bf16* awh     = bafl + (size_t)B_ * A_ * D_;            // D_*ATWP each
    __bf16* awl     = awh  + (size_t)D_ * ATWP;
    __bf16* hwh     = awl  + (size_t)D_ * ATWP;               // D2*HWS each
    __bf16* hwl     = hwh  + (size_t)D2 * HWS;

    k_conv1x1<<<(B_ * CF * FH * FW) / 256, 256, 0, stream>>>(x, conv_w, conv_b, feats);
    k_split_attn_w<<<(D_ * ATWP) / 256, 256, 0, stream>>>(attn_w, awh, awl);
    k_prep_head<<<(D2 * HWS) / 256, 256, 0, stream>>>(cls_w, cls_b, reg_w, reg_b, hwh, hwl, headB);
    k_gather<<<(int)(((size_t)B_ * A_ * D_) / 256), 256, 0, stream>>>(feats, cut_xs, invalid_mask, bafh, bafl);
    k_gemm_scores<<<(B_ * MT1 * NG1) / 8, 256, 0, stream>>>(bafh, bafl, awh, awl, attn_b, attn);
    k_softmax<<<B_ * A_, 256, 0, stream>>>(attn);
    k_gemm_att<<<(B_ * MT1 * NG2) / 8, 256, 0, stream>>>(attn, bafh, bafl, att_feat);
    k_gemm_head<<<(MTH * NGH) / 8, 256, 0, stream>>>(att_feat, bafh, bafl, hwh, hwl, headB, head_out);
    k_props<<<(B_ * A_ * P_ + 255) / 256, 256, 0, stream>>>(head_out, anchors, props);
}
